// PIGNN_85555748537205
// MI455X (gfx1250) — matrix operand order repaired and checked by reading, not compiled, not run on hardware
//
#include <hip/hip_runtime.h>

#define NN 20000
#define NE 320000
#define HDIM 128
#define NL 6
#define LDA 136   // LDS row stride in bf16 elems (128 + 8 pad), keeps 16B alignment

typedef __attribute__((ext_vector_type(16))) __bf16 v16bf;
typedef __attribute__((ext_vector_type(8)))  float  v8f;

struct __align__(16) BF16x8 { unsigned short h[8]; };
struct __align__(16) F32x4  { float x, y, z, w; };

union V16BF { v16bf v; BF16x8 q[2]; };

__device__ __forceinline__ unsigned short f2bf(float x) {
  unsigned int u = __float_as_uint(x);
  unsigned int r = (u + 0x7FFFu + ((u >> 16) & 1u)) >> 16;
  return (unsigned short)r;
}

__device__ __forceinline__ v8f wmma_bf16(const V16BF& a, const V16BF& b, v8f c) {
  return __builtin_amdgcn_wmma_f32_16x16x32_bf16(false, a.v, false, b.v,
                                                 (short)0, c, false, false);
}

// ---------------------------------------------------------------------------
// Weight transpose + f32->bf16:  src (L,K,N) row-major  ->  dst (L,N,K)
// ---------------------------------------------------------------------------
__global__ void wt_t_bf16_kernel(const float* __restrict__ src,
                                 unsigned short* __restrict__ dst,
                                 int L, int K, int N, int total) {
  int i = blockIdx.x * blockDim.x + threadIdx.x;
  if (i >= total) return;
  int kn = K * N;
  int l = i / kn;
  int rem = i - l * kn;
  int k = rem / N;
  int n = rem - k * N;
  dst[(size_t)l * kn + (size_t)n * K + k] = f2bf(src[i]);
}

// ---------------------------------------------------------------------------
// Node encoder: h = relu(x@W1+b1)@W2+b2   (coords x[:,0:3] forced to 0)
// ---------------------------------------------------------------------------
__global__ __launch_bounds__(128) void node_enc_kernel(
    const float* __restrict__ x,
    const float* __restrict__ w1, const float* __restrict__ b1,
    const float* __restrict__ w2, const float* __restrict__ b2,
    float* __restrict__ h, unsigned short* __restrict__ hb) {
  __shared__ float xr[9];
  __shared__ float t[HDIM];
  const int node = blockIdx.x, c = threadIdx.x;
  if (c < 9) xr[c] = (c < 3) ? 0.f : x[(size_t)node * 9 + c];
  __syncthreads();
  float a = b1[c];
  #pragma unroll
  for (int k = 3; k < 9; ++k) a += xr[k] * w1[k * HDIM + c];
  t[c] = a > 0.f ? a : 0.f;
  __syncthreads();
  float a2 = b2[c];
  for (int k = 0; k < HDIM; ++k) a2 += t[k] * w2[k * HDIM + c];
  h[(size_t)node * HDIM + c] = a2;
  hb[(size_t)node * HDIM + c] = f2bf(a2);
}

// ---------------------------------------------------------------------------
// Edge encoder: e = relu(ea@W1+b1)@W2+b2  (bf16 output only; e is constant)
// ---------------------------------------------------------------------------
__global__ __launch_bounds__(128) void edge_enc_kernel(
    const float* __restrict__ ea,
    const float* __restrict__ w1, const float* __restrict__ b1,
    const float* __restrict__ w2, const float* __restrict__ b2,
    unsigned short* __restrict__ eb) {
  __shared__ float er[10];
  __shared__ float t[HDIM];
  const int e = blockIdx.x, c = threadIdx.x;
  if (c < 10) er[c] = ea[(size_t)e * 10 + c];
  __syncthreads();
  float a = b1[c];
  #pragma unroll
  for (int k = 0; k < 10; ++k) a += er[k] * w1[k * HDIM + c];
  t[c] = a > 0.f ? a : 0.f;
  __syncthreads();
  float a2 = b2[c];
  for (int k = 0; k < HDIM; ++k) a2 += t[k] * w2[k * HDIM + c];
  eb[(size_t)e * HDIM + c] = f2bf(a2);
}

// ---------------------------------------------------------------------------
// Edge message kernel (WMMA): m = relu([h_src|h_dst|e]@W1+b1)@W2+b2,
// scatter-add into agg[dst]. 256 threads = 8 waves, 128 edges per block.
// Wave w owns output columns [16w,16w+16); loops 8 row tiles.
// ---------------------------------------------------------------------------
__global__ __launch_bounds__(256) void edge_mp_kernel(
    const unsigned short* __restrict__ hb,
    const unsigned short* __restrict__ eb,
    const int* __restrict__ ei,
    const unsigned short* __restrict__ w1t, const float* __restrict__ b1,
    const unsigned short* __restrict__ w2t, const float* __restrict__ b2,
    float* __restrict__ agg) {
  __shared__ __align__(16) unsigned short Abuf[128 * LDA];
  __shared__ int ssrc[128];
  __shared__ int sdst[128];

  const int tid = threadIdx.x;
  const int lane = tid & 31;
  const int wave = tid >> 5;
  const int hi = lane >> 4;     // half-wave select (WMMA A/B K-group)
  const int ln = lane & 15;     // row (A) / col (B,D) within tile
  const int e0 = blockIdx.x * 128;
  const int srow = tid >> 1;    // staging: 2 threads per row
  const int shalf = tid & 1;

  if (tid < 128) {
    ssrc[tid] = ei[e0 + tid];
    sdst[tid] = ei[NE + e0 + tid];
  }
  __builtin_prefetch(w1t, 0, 3);
  __builtin_prefetch(w2t, 0, 3);
  __syncthreads();

  const v8f vzero = {0.f, 0.f, 0.f, 0.f, 0.f, 0.f, 0.f, 0.f};
  v8f acc[8];
  #pragma unroll
  for (int r = 0; r < 8; ++r) acc[r] = vzero;

  // ---- GEMM1 over the 3 concat segments (each exactly 128 wide) ----
  for (int seg = 0; seg < 3; ++seg) {
    {
      const unsigned short* src;
      if (seg == 0)      src = hb + (size_t)ssrc[srow] * HDIM + shalf * 64;
      else if (seg == 1) src = hb + (size_t)sdst[srow] * HDIM + shalf * 64;
      else               src = eb + (size_t)(e0 + srow) * HDIM + shalf * 64;
      const BF16x8* s8 = (const BF16x8*)src;
      BF16x8* d8 = (BF16x8*)(Abuf + srow * LDA + shalf * 64);
      #pragma unroll
      for (int i = 0; i < 8; ++i) d8[i] = s8[i];
    }
    __syncthreads();
    #pragma unroll
    for (int kk = 0; kk < 4; ++kk) {
      V16BF bf;
      const unsigned short* bp =
          w1t + (size_t)(wave * 16 + ln) * (3 * HDIM) + seg * HDIM + kk * 32 + hi * 8;
      bf.q[0] = *(const BF16x8*)(bp);
      bf.q[1] = *(const BF16x8*)(bp + 16);
      #pragma unroll
      for (int r = 0; r < 8; ++r) {
        V16BF af;
        const unsigned short* ap = Abuf + (r * 16 + ln) * LDA + kk * 32 + hi * 8;
        af.q[0] = *(const BF16x8*)(ap);
        af.q[1] = *(const BF16x8*)(ap + 16);
        acc[r] = wmma_bf16(af, bf, acc[r]);
      }
    }
    __syncthreads();
  }

  // ---- bias + ReLU -> bf16 hidden staged back into Abuf ----
  {
    const float bias = b1[wave * 16 + ln];
    #pragma unroll
    for (int r = 0; r < 8; ++r) {
      #pragma unroll
      for (int v = 0; v < 8; ++v) {
        float val = acc[r][v] + bias;
        val = val > 0.f ? val : 0.f;
        Abuf[(r * 16 + v + 8 * hi) * LDA + wave * 16 + ln] = f2bf(val);
      }
      acc[r] = vzero;
    }
  }
  __syncthreads();

  // ---- GEMM2: m = hidden @ W2 ----
  #pragma unroll
  for (int kk = 0; kk < 4; ++kk) {
    V16BF bf;
    const unsigned short* bp =
        w2t + (size_t)(wave * 16 + ln) * HDIM + kk * 32 + hi * 8;
    bf.q[0] = *(const BF16x8*)(bp);
    bf.q[1] = *(const BF16x8*)(bp + 16);
    #pragma unroll
    for (int r = 0; r < 8; ++r) {
      V16BF af;
      const unsigned short* ap = Abuf + (r * 16 + ln) * LDA + kk * 32 + hi * 8;
      af.q[0] = *(const BF16x8*)(ap);
      af.q[1] = *(const BF16x8*)(ap + 16);
      acc[r] = wmma_bf16(af, bf, acc[r]);
    }
  }

  // ---- scatter-add (segment_sum over dst) ----
  {
    const float bias = b2[wave * 16 + ln];
    const int col = wave * 16 + ln;
    #pragma unroll
    for (int r = 0; r < 8; ++r) {
      #pragma unroll
      for (int v = 0; v < 8; ++v) {
        const int row = r * 16 + v + 8 * hi;
        const int d = sdst[row];
        atomicAdd(&agg[(size_t)d * HDIM + col], acc[r][v] + bias);
      }
    }
  }
}

// ---------------------------------------------------------------------------
// Node update kernel (WMMA): upd = relu([h|agg]@W1+b1)@W2+b2 ; h += upd
// ---------------------------------------------------------------------------
__global__ __launch_bounds__(256) void node_mp_kernel(
    const unsigned short* __restrict__ w1t, const float* __restrict__ b1,
    const unsigned short* __restrict__ w2t, const float* __restrict__ b2,
    const float* __restrict__ agg,
    float* h, unsigned short* hb) {
  __shared__ __align__(16) unsigned short Abuf[128 * LDA];

  const int tid = threadIdx.x;
  const int lane = tid & 31;
  const int wave = tid >> 5;
  const int hi = lane >> 4;
  const int ln = lane & 15;
  const int base = blockIdx.x * 128;
  const int srow = tid >> 1;
  const int shalf = tid & 1;
  const int snode = base + srow;

  __builtin_prefetch(w1t, 0, 3);
  __builtin_prefetch(w2t, 0, 3);

  const v8f vzero = {0.f, 0.f, 0.f, 0.f, 0.f, 0.f, 0.f, 0.f};
  v8f acc[8];
  #pragma unroll
  for (int r = 0; r < 8; ++r) acc[r] = vzero;

  for (int seg = 0; seg < 2; ++seg) {
    {
      unsigned short* dst = Abuf + srow * LDA + shalf * 64;
      if (snode < NN) {
        if (seg == 0) {
          const BF16x8* s8 = (const BF16x8*)(hb + (size_t)snode * HDIM + shalf * 64);
          BF16x8* d8 = (BF16x8*)dst;
          #pragma unroll
          for (int i = 0; i < 8; ++i) d8[i] = s8[i];
        } else {
          const F32x4* s4 = (const F32x4*)(agg + (size_t)snode * HDIM + shalf * 64);
          #pragma unroll
          for (int i = 0; i < 16; ++i) {
            F32x4 f = s4[i];
            dst[i * 4 + 0] = f2bf(f.x);
            dst[i * 4 + 1] = f2bf(f.y);
            dst[i * 4 + 2] = f2bf(f.z);
            dst[i * 4 + 3] = f2bf(f.w);
          }
        }
      } else {
        #pragma unroll
        for (int i = 0; i < 64; ++i) dst[i] = 0;
      }
    }
    __syncthreads();
    #pragma unroll
    for (int kk = 0; kk < 4; ++kk) {
      V16BF bf;
      const unsigned short* bp =
          w1t + (size_t)(wave * 16 + ln) * (2 * HDIM) + seg * HDIM + kk * 32 + hi * 8;
      bf.q[0] = *(const BF16x8*)(bp);
      bf.q[1] = *(const BF16x8*)(bp + 16);
      #pragma unroll
      for (int r = 0; r < 8; ++r) {
        V16BF af;
        const unsigned short* ap = Abuf + (r * 16 + ln) * LDA + kk * 32 + hi * 8;
        af.q[0] = *(const BF16x8*)(ap);
        af.q[1] = *(const BF16x8*)(ap + 16);
        acc[r] = wmma_bf16(af, bf, acc[r]);
      }
    }
    __syncthreads();
  }

  {
    const float bias = b1[wave * 16 + ln];
    #pragma unroll
    for (int r = 0; r < 8; ++r) {
      #pragma unroll
      for (int v = 0; v < 8; ++v) {
        float val = acc[r][v] + bias;
        val = val > 0.f ? val : 0.f;
        Abuf[(r * 16 + v + 8 * hi) * LDA + wave * 16 + ln] = f2bf(val);
      }
      acc[r] = vzero;
    }
  }
  __syncthreads();

  #pragma unroll
  for (int kk = 0; kk < 4; ++kk) {
    V16BF bf;
    const unsigned short* bp =
        w2t + (size_t)(wave * 16 + ln) * HDIM + kk * 32 + hi * 8;
    bf.q[0] = *(const BF16x8*)(bp);
    bf.q[1] = *(const BF16x8*)(bp + 16);
    #pragma unroll
    for (int r = 0; r < 8; ++r) {
      V16BF af;
      const unsigned short* ap = Abuf + (r * 16 + ln) * LDA + kk * 32 + hi * 8;
      af.q[0] = *(const BF16x8*)(ap);
      af.q[1] = *(const BF16x8*)(ap + 16);
      acc[r] = wmma_bf16(af, bf, acc[r]);
    }
  }

  {
    const float bias = b2[wave * 16 + ln];
    const int col = wave * 16 + ln;
    #pragma unroll
    for (int r = 0; r < 8; ++r) {
      #pragma unroll
      for (int v = 0; v < 8; ++v) {
        const int row = r * 16 + v + 8 * hi;
        const int node = base + row;
        if (node < NN) {
          float val = h[(size_t)node * HDIM + col] + acc[r][v] + bias;
          h[(size_t)node * HDIM + col] = val;
          hb[(size_t)node * HDIM + col] = f2bf(val);
        }
      }
    }
  }
}

// ---------------------------------------------------------------------------
// Decoder + BC masks: pred = relu(relu(h@W1+b1)@W2+b2)@W3+b3, masked
// ---------------------------------------------------------------------------
__global__ __launch_bounds__(128) void decoder_kernel(
    const float* __restrict__ h,
    const float* __restrict__ w1, const float* __restrict__ b1,
    const float* __restrict__ w2, const float* __restrict__ b2,
    const float* __restrict__ w3, const float* __restrict__ b3,
    const float* __restrict__ bc_disp, const float* __restrict__ bc_rot,
    const float* __restrict__ fmask,
    float* __restrict__ out) {
  __shared__ float hrow[HDIM];
  __shared__ float t1[HDIM];
  __shared__ float t2[64];
  const int node = blockIdx.x, c = threadIdx.x;
  hrow[c] = h[(size_t)node * HDIM + c];
  __syncthreads();
  float a = b1[c];
  for (int k = 0; k < HDIM; ++k) a += hrow[k] * w1[k * HDIM + c];
  t1[c] = a > 0.f ? a : 0.f;
  __syncthreads();
  if (c < 64) {
    float a2 = b2[c];
    for (int k = 0; k < HDIM; ++k) a2 += t1[k] * w2[k * 64 + c];
    t2[c] = a2 > 0.f ? a2 : 0.f;
  }
  __syncthreads();
  if (c < 15) {
    float p = b3[c];
    #pragma unroll
    for (int k = 0; k < 64; ++k) p += t2[k] * w3[k * 15 + c];
    float m;
    if (c < 2)       m = 1.f - bc_disp[node];
    else if (c == 2) m = 1.f - bc_rot[node];
    else             m = fmask[node * 4 + (c - 3) / 3];
    out[(size_t)node * 15 + c] = p * m;
  }
}

// ---------------------------------------------------------------------------
extern "C" void kernel_launch(void* const* d_in, const int* in_sizes, int n_in,
                              void* d_out, int out_size, void* d_ws, size_t ws_size,
                              hipStream_t stream) {
  (void)in_sizes; (void)n_in; (void)out_size; (void)ws_size;

  const float* x        = (const float*)d_in[0];
  const float* ea       = (const float*)d_in[1];
  const float* bc_disp  = (const float*)d_in[2];
  const float* bc_rot   = (const float*)d_in[3];
  const float* fmask    = (const float*)d_in[4];
  const int*   ei       = (const int*)  d_in[5];
  const float* enc_n_w1 = (const float*)d_in[6];
  const float* enc_n_b1 = (const float*)d_in[7];
  const float* enc_n_w2 = (const float*)d_in[8];
  const float* enc_n_b2 = (const float*)d_in[9];
  const float* enc_e_w1 = (const float*)d_in[10];
  const float* enc_e_b1 = (const float*)d_in[11];
  const float* enc_e_w2 = (const float*)d_in[12];
  const float* enc_e_b2 = (const float*)d_in[13];
  const float* mp_e_w1  = (const float*)d_in[14];
  const float* mp_e_b1  = (const float*)d_in[15];
  const float* mp_e_w2  = (const float*)d_in[16];
  const float* mp_e_b2  = (const float*)d_in[17];
  const float* mp_n_w1  = (const float*)d_in[18];
  const float* mp_n_b1  = (const float*)d_in[19];
  const float* mp_n_w2  = (const float*)d_in[20];
  const float* mp_n_b2  = (const float*)d_in[21];
  const float* dec_w1   = (const float*)d_in[22];
  const float* dec_b1   = (const float*)d_in[23];
  const float* dec_w2   = (const float*)d_in[24];
  const float* dec_b2   = (const float*)d_in[25];
  const float* dec_w3   = (const float*)d_in[26];
  const float* dec_b3   = (const float*)d_in[27];

  // Workspace layout
  char* p = (char*)d_ws;
  auto carve = [&](size_t bytes) -> void* {
    void* r = (void*)p;
    p += (bytes + 255) & ~(size_t)255;
    return r;
  };
  float*          h    = (float*)carve((size_t)NN * HDIM * 4);
  unsigned short* hb   = (unsigned short*)carve((size_t)NN * HDIM * 2);
  unsigned short* eb   = (unsigned short*)carve((size_t)NE * HDIM * 2);
  float*          agg  = (float*)carve((size_t)NN * HDIM * 4);
  unsigned short* we1t = (unsigned short*)carve((size_t)NL * 3 * HDIM * HDIM * 2);
  unsigned short* we2t = (unsigned short*)carve((size_t)NL * HDIM * HDIM * 2);
  unsigned short* wn1t = (unsigned short*)carve((size_t)NL * 2 * HDIM * HDIM * 2);
  unsigned short* wn2t = (unsigned short*)carve((size_t)NL * HDIM * HDIM * 2);

  // Transpose + bf16-convert MP weights
  {
    int tot = NL * 3 * HDIM * HDIM;
    wt_t_bf16_kernel<<<(tot + 255) / 256, 256, 0, stream>>>(mp_e_w1, we1t, NL, 3 * HDIM, HDIM, tot);
    tot = NL * HDIM * HDIM;
    wt_t_bf16_kernel<<<(tot + 255) / 256, 256, 0, stream>>>(mp_e_w2, we2t, NL, HDIM, HDIM, tot);
    tot = NL * 2 * HDIM * HDIM;
    wt_t_bf16_kernel<<<(tot + 255) / 256, 256, 0, stream>>>(mp_n_w1, wn1t, NL, 2 * HDIM, HDIM, tot);
    tot = NL * HDIM * HDIM;
    wt_t_bf16_kernel<<<(tot + 255) / 256, 256, 0, stream>>>(mp_n_w2, wn2t, NL, HDIM, HDIM, tot);
  }

  // Encoders
  node_enc_kernel<<<NN, 128, 0, stream>>>(x, enc_n_w1, enc_n_b1, enc_n_w2, enc_n_b2, h, hb);
  edge_enc_kernel<<<NE, 128, 0, stream>>>(ea, enc_e_w1, enc_e_b1, enc_e_w2, enc_e_b2, eb);

  // Message passing
  for (int l = 0; l < NL; ++l) {
    hipMemsetAsync(agg, 0, (size_t)NN * HDIM * 4, stream);
    edge_mp_kernel<<<NE / 128, 256, 0, stream>>>(
        hb, eb, ei,
        we1t + (size_t)l * 3 * HDIM * HDIM, mp_e_b1 + l * HDIM,
        we2t + (size_t)l * HDIM * HDIM,     mp_e_b2 + l * HDIM,
        agg);
    node_mp_kernel<<<(NN + 127) / 128, 256, 0, stream>>>(
        wn1t + (size_t)l * 2 * HDIM * HDIM, mp_n_b1 + l * HDIM,
        wn2t + (size_t)l * HDIM * HDIM,     mp_n_b2 + l * HDIM,
        agg, h, hb);
  }

  // Decoder + masks
  decoder_kernel<<<NN, 128, 0, stream>>>(h, dec_w1, dec_b1, dec_w2, dec_b2,
                                         dec_w3, dec_b3, bc_disp, bc_rot, fmask,
                                         (float*)d_out);
}
